// Attention_39651138077091
// MI455X (gfx1250) — matrix operand order repaired and checked by reading, not compiled, not run on hardware
//
#include <hip/hip_runtime.h>
#include <math.h>

// Shapes (fixed by the reference)
#define B_   32
#define TX_  2048
#define DH_  512
#define U2_  1024

typedef __attribute__((ext_vector_type(16))) __bf16 v16bf;
typedef __attribute__((ext_vector_type(8)))  float  v8f;

// LDS layout for the WMMA kernel (bytes)
#define SA_BYTES   (64 * U2_ * 2)          // 128 KB : A tile, 64 rows x 1024 bf16
#define SB_BYTES   (32 * U2_ * 2)          // 64 KB  : one B pair-tile (32 cols x 1024 K)
#define SE_OFF     (SA_BYTES + 2 * SB_BYTES)
#define SMEM_BYTES (SE_OFF + 128 * 4)      // + reduction scratch

// ---------------------------------------------------------------------------
// K0: Wa (U2 x U2 f32, row-major [k][n]) -> WaT bf16 row-major [n][k]
// ---------------------------------------------------------------------------
__global__ __launch_bounds__(256)
void wa_transpose_bf16_kernel(const float* __restrict__ Wa, __bf16* __restrict__ WaT)
{
    __shared__ float tile[32][33];
    const int tx = threadIdx.x;         // 0..31
    const int ty = threadIdx.y;         // 0..7
    const int nb = blockIdx.x * 32;     // n block
    const int kb = blockIdx.y * 32;     // k block
    for (int j = 0; j < 4; ++j) {
        int k = kb + ty + j * 8;
        tile[ty + j * 8][tx] = Wa[(size_t)k * U2_ + (nb + tx)];
    }
    __syncthreads();
    for (int j = 0; j < 4; ++j) {
        int n = nb + ty + j * 8;
        WaT[(size_t)n * U2_ + (kb + tx)] = (__bf16)tile[tx][ty + j * 8];
    }
}

// ---------------------------------------------------------------------------
// K1: dec[b][e] = h[b,:] @ Wh_w[:,e] + Wh_b[e]
// ---------------------------------------------------------------------------
__global__ __launch_bounds__(256)
void dec_kernel(const float* __restrict__ h, const float* __restrict__ Whw,
                const float* __restrict__ Whb, float* __restrict__ dec)
{
    const int e = blockIdx.x * 256 + threadIdx.x;   // 0..1023
    const int b = blockIdx.y;                       // 0..31
    float acc = Whb[e];
    const float* hb = h + (size_t)b * DH_;
    for (int d = 0; d < DH_; ++d)
        acc += hb[d] * Whw[(size_t)d * U2_ + e];
    dec[(size_t)b * U2_ + e] = acc;
}

// ---------------------------------------------------------------------------
// K2: fused  e[b,t] = v . tanh(a[b,t,:] @ Wa + dec[b,:] + cov[b,t]*Wc)
// Block: 256 threads (8 waves). Owns 64 consecutive rows (one b).
//  - wave w: row-tile (w&3), column-half (w>>2) of the current 32-col stripe
//  - B stripes double-buffered in LDS, filled with async global->LDS DMA
//    (GLOBAL_LOAD_ASYNC_TO_LDS_B128, ASYNCcnt) overlapping the K loop.
// ---------------------------------------------------------------------------
union Frag { v16bf v; uint4 q[2]; };

__device__ __forceinline__ void async_load_b_pair(const __bf16* __restrict__ WaT,
                                                  int pair, char* sBbuf, int tid)
{
    // 32 rows x 1024 bf16 = 64KB = 4096 x b128; 16 per thread
    const unsigned lbase = (unsigned)(uintptr_t)sBbuf;      // LDS offset (addr[31:0])
    const unsigned long long gbase =
        (unsigned long long)(uintptr_t)(WaT + (size_t)pair * 32 * U2_);
#pragma unroll
    for (int j = 0; j < 16; ++j) {
        const int i = tid + j * 256;
        const unsigned la = lbase + (unsigned)i * 16u;
        const unsigned long long ga = gbase + (unsigned long long)i * 16ull;
        asm volatile("global_load_async_to_lds_b128 %0, %1, off"
                     :: "v"(la), "v"(ga) : "memory");
    }
}

__global__ __launch_bounds__(256)
void attn_energy_kernel(const float* __restrict__ a, const float* __restrict__ cov,
                        const float* __restrict__ dec, const __bf16* __restrict__ WaT,
                        const float* __restrict__ Wc, const float* __restrict__ vvec,
                        const int* __restrict__ use_cov, float* __restrict__ e_out)
{
    extern __shared__ __align__(16) char smem[];
    __bf16* sA  = (__bf16*)smem;                    // [64][1024]
    char*   sB0 = smem + SA_BYTES;                  // [32][1024] bf16
    char*   sB1 = smem + SA_BYTES + SB_BYTES;       // [32][1024] bf16
    float*  sE  = (float*)(smem + SE_OFF);          // [2][64]

    const int tid  = threadIdx.x;
    const int wave = tid >> 5;
    const int lane = tid & 31;
    const long rowBase = (long)blockIdx.x * 64;     // global row (b*TX + t)
    const int  b     = (int)(rowBase >> 11);        // TX_ = 2048
    const int  tBase = (int)(rowBase & (TX_ - 1));

    // ---- stage A: 64 rows x 1024 f32 -> bf16 in LDS ----
    for (int idx = tid; idx < 64 * 256; idx += 256) {
        const int r  = idx >> 8;
        const int c4 = (idx & 255) << 2;
        const float4 f = *(const float4*)(a + (rowBase + r) * (size_t)U2_ + c4);
        union { __bf16 h[4]; uint2 u; } p;
        p.h[0] = (__bf16)f.x; p.h[1] = (__bf16)f.y;
        p.h[2] = (__bf16)f.z; p.h[3] = (__bf16)f.w;
        *(uint2*)(sA + (size_t)r * U2_ + c4) = p.u;
    }

    // Fragment geometry (ISA 7.12.2: 16-bit A 16x32, 16-bit B 32x16, f32 C 16x16)
    const int mA      = lane & 15;        // A: row within tile
    const int khA     = lane >> 4;        // A: K half selector
    const int colC    = lane & 15;        // B/C: column within tile
    const int halfC   = lane >> 4;        // B/C: half selector
    const int rowTile = wave & 3;         // which 16-row tile of the 64
    const int nHalf   = wave >> 2;        // which 16-col half of the 32-col stripe
    const int waveRow = rowTile << 4;

    const int useCov = use_cov[0];
    float covRow[8];
    for (int r = 0; r < 8; ++r) {
        const int m = waveRow + r + (halfC << 3);
        covRow[r] = useCov ? cov[(size_t)b * TX_ + tBase + m] : 0.0f;
    }

    float eacc[8] = {0.f, 0.f, 0.f, 0.f, 0.f, 0.f, 0.f, 0.f};

    // prologue: DMA stripe 0 into sB0, close the A-staging too
    async_load_b_pair(WaT, 0, sB0, tid);
    asm volatile("s_wait_asynccnt 0x0" ::: "memory");
    __syncthreads();

    const __bf16* paBase = sA + (size_t)(waveRow + mA) * U2_ + (khA << 3);

#pragma unroll 1
    for (int p = 0; p < U2_ / 32; ++p) {
        char* curB = (p & 1) ? sB1 : sB0;
        char* nxtB = (p & 1) ? sB0 : sB1;
        if (p + 1 < U2_ / 32)
            async_load_b_pair(WaT, p + 1, nxtB, tid);   // overlaps the K loop below

        const __bf16* pbBase = (const __bf16*)curB
                             + (size_t)(nHalf * 16 + colC) * U2_ + (halfC << 3);

        // ---- K loop: software-pipelined ds_load_b128 fragments + WMMA ----
        Frag fa[2], fb[2];
        fa[0].q[0] = *(const uint4*)(paBase);
        fa[0].q[1] = *(const uint4*)(paBase + 16);
        fb[0].q[0] = *(const uint4*)(pbBase);
        fb[0].q[1] = *(const uint4*)(pbBase + 16);
        v8f c = {};
#pragma unroll
        for (int ks = 0; ks < 32; ++ks) {
            const int cb = ks & 1;
            if (ks + 1 < 32) {
                const __bf16* pa = paBase + (ks + 1) * 32;
                const __bf16* pb = pbBase + (ks + 1) * 32;
                fa[cb ^ 1].q[0] = *(const uint4*)(pa);
                fa[cb ^ 1].q[1] = *(const uint4*)(pa + 16);
                fb[cb ^ 1].q[0] = *(const uint4*)(pb);
                fb[cb ^ 1].q[1] = *(const uint4*)(pb + 16);
            }
            c = __builtin_amdgcn_wmma_f32_16x16x32_bf16(
                    false, fa[cb].v, false, fb[cb].v,
                    (short)0, c, false, false);
        }

        // ---- epilogue for this 16-col tile of this wave ----
        const int n16 = p * 32 + nHalf * 16 + colC;
        const float dv  = dec[(size_t)b * U2_ + n16];
        const float wcv = Wc[n16];
        const float vv  = vvec[n16];
#pragma unroll
        for (int r = 0; r < 8; ++r) {
            const float f = c[r] + dv + covRow[r] * wcv;
            eacc[r] += tanhf(f) * vv;
        }

        asm volatile("s_wait_asynccnt 0x0" ::: "memory");
        __syncthreads();   // next stripe resident; everyone done with curB
    }

    // reduce the 16 per-column partials across each 16-lane half (wave32)
#pragma unroll
    for (int off = 8; off; off >>= 1)
#pragma unroll
        for (int r = 0; r < 8; ++r)
            eacc[r] += __shfl_xor(eacc[r], off, 16);

    if ((lane & 15) == 0) {
#pragma unroll
        for (int r = 0; r < 8; ++r) {
            const int m = waveRow + r + (halfC << 3);
            sE[nHalf * 64 + m] = eacc[r];       // two column-half partials per row
        }
    }
    __syncthreads();
    if (tid < 64)
        e_out[rowBase + tid] = sE[tid] + sE[64 + tid];
}

// ---------------------------------------------------------------------------
// K3: per-batch masked softmax + renormalize -> alpha
// ---------------------------------------------------------------------------
__global__ __launch_bounds__(256)
void softmax_mask_kernel(const float* __restrict__ e, const unsigned char* __restrict__ mask,
                         const int* __restrict__ use_mask, float* __restrict__ alpha)
{
    __shared__ float red[256];
    const int b = blockIdx.x;
    const int tid = threadIdx.x;
    const float* eb = e + (size_t)b * TX_;

    float ev[8];
    float m = -3.402823466e+38f;
    for (int i = 0; i < 8; ++i) {
        ev[i] = eb[tid + i * 256];
        m = fmaxf(m, ev[i]);
    }
    red[tid] = m; __syncthreads();
    for (int s = 128; s; s >>= 1) {
        if (tid < s) red[tid] = fmaxf(red[tid], red[tid + s]);
        __syncthreads();
    }
    const float M = red[0]; __syncthreads();

    float s0 = 0.f;
    for (int i = 0; i < 8; ++i) { ev[i] = __expf(ev[i] - M); s0 += ev[i]; }
    red[tid] = s0; __syncthreads();
    for (int s = 128; s; s >>= 1) {
        if (tid < s) red[tid] += red[tid + s];
        __syncthreads();
    }
    const float S = red[0]; __syncthreads();

    const int um = use_mask[0];
    float s1 = 0.f;
    for (int i = 0; i < 8; ++i) {
        float al = ev[i] / S;
        if (um && mask[(size_t)b * TX_ + tid + i * 256] == 0) al = 0.f;
        ev[i] = al; s1 += al;
    }
    red[tid] = s1; __syncthreads();
    for (int s = 128; s; s >>= 1) {
        if (tid < s) red[tid] += red[tid + s];
        __syncthreads();
    }
    const float S2 = um ? red[0] : 1.0f;

    for (int i = 0; i < 8; ++i)
        alpha[(size_t)b * TX_ + tid + i * 256] = ev[i] / S2;
}

// ---------------------------------------------------------------------------
// K4: context[b,e] = sum_t alpha[b,t] * a[b,t,e]
// ---------------------------------------------------------------------------
__global__ __launch_bounds__(256)
void context_kernel(const float* __restrict__ a, const float* __restrict__ alpha,
                    float* __restrict__ ctx)
{
    __shared__ float sal[TX_];
    const int tid = threadIdx.x;
    const int b = blockIdx.y;
    const int e = blockIdx.x * 256 + tid;
    for (int t = tid; t < TX_; t += 256) sal[t] = alpha[(size_t)b * TX_ + t];
    __syncthreads();
    float acc = 0.f;
    const float* ab = a + (size_t)b * TX_ * U2_ + e;
    for (int t = 0; t < TX_; ++t)
        acc += sal[t] * ab[(size_t)t * U2_];
    ctx[(size_t)b * U2_ + e] = acc;
}

// ---------------------------------------------------------------------------
// Launcher
// ---------------------------------------------------------------------------
extern "C" void kernel_launch(void* const* d_in, const int* in_sizes, int n_in,
                              void* d_out, int out_size, void* d_ws, size_t ws_size,
                              hipStream_t stream)
{
    (void)in_sizes; (void)n_in; (void)out_size; (void)ws_size;
    const float*         a        = (const float*)d_in[0];
    const float*         h        = (const float*)d_in[1];
    const float*         cov      = (const float*)d_in[2];
    const unsigned char* X_mask   = (const unsigned char*)d_in[3];
    const float*         Wa       = (const float*)d_in[4];
    const float*         Wh_w     = (const float*)d_in[5];
    const float*         Wh_b     = (const float*)d_in[6];
    const float*         Wc       = (const float*)d_in[7];
    const float*         v        = (const float*)d_in[8];
    const int*           use_cov  = (const int*)d_in[9];
    const int*           use_mask = (const int*)d_in[10];

    // workspace layout
    char* ws = (char*)d_ws;
    __bf16* WaT  = (__bf16*)ws;                                  // 2 MB
    float*  dec  = (float*)(ws + (size_t)U2_ * U2_ * 2);         // 128 KB
    float*  e    = (float*)(ws + (size_t)U2_ * U2_ * 2 + (size_t)B_ * U2_ * 4); // 256 KB

    float* ctx_out   = (float*)d_out;            // B*U2 = 32768 floats
    float* alpha_out = (float*)d_out + B_ * U2_; // B*TX = 65536 floats

    // K0: Wa -> WaT (bf16, transposed)
    {
        dim3 grid(U2_ / 32, U2_ / 32), block(32, 8);
        wa_transpose_bf16_kernel<<<grid, block, 0, stream>>>(Wa, WaT);
    }
    // K1: dec
    {
        dim3 grid(U2_ / 256, B_), block(256);
        dec_kernel<<<grid, block, 0, stream>>>(h, Wh_w, Wh_b, dec);
    }
    // K2: fused WMMA GEMM + tanh + dot(v) -> e
    {
        const int nBlocks = (B_ * TX_) / 64;            // 1024
        attn_energy_kernel<<<nBlocks, 256, SMEM_BYTES, stream>>>(a, cov, dec, WaT,
                                                                 Wc, v, use_cov, e);
    }
    // K3: masked softmax -> alpha
    softmax_mask_kernel<<<B_, 256, 0, stream>>>(e, X_mask, use_mask, alpha_out);
    // K4: context
    {
        dim3 grid(U2_ / 256, B_), block(256);
        context_kernel<<<grid, block, 0, stream>>>(a, alpha_out, ctx_out);
    }
}
